// YOLOXLoss_44805098831942
// MI455X (gfx1250) — compile-verified, matrix operand order checked by reading, not run on hardware
//
#include <hip/hip_runtime.h>
#include <math.h>

#define A_TOT   8400
#define NCLS    80
#define MGT     50
#define TOPKN   10
#define BIGC    1e15f
#define CHUNK   64
#define NCHUNK  ((A_TOT + CHUNK - 1) / CHUNK)   /* 132 */
#define NXB     ((A_TOT + 255) / 256)           /* 33  */
#define PI_F    3.14159265358979323846f

typedef __attribute__((ext_vector_type(16))) _Float16 v16h;
typedef __attribute__((ext_vector_type(8)))  float    v8f;

#if defined(__gfx1250__) && __has_builtin(__builtin_amdgcn_global_load_async_to_lds_b32) && __has_builtin(__builtin_amdgcn_s_wait_asynccnt)
#define USE_ASYNC_LDS 1
#else
#define USE_ASYNC_LDS 0
#endif

__device__ __forceinline__ float sigf(float x) { return 1.f / (1.f + expf(-x)); }
__device__ __forceinline__ float bcef(float x, float t) {
    return fmaxf(x, 0.f) - x * t + log1pf(expf(-fabsf(x)));
}
__device__ __forceinline__ void anchor_of(int a, float& xc, float& yc, float& st) {
    int gx, gy;
    if (a < 6400)      { st =  8.f; gx = a % 80; gy = a / 80; }
    else if (a < 8000) { int i = a - 6400; st = 16.f; gx = i % 40; gy = i / 40; }
    else               { int i = a - 8000; st = 32.f; gx = i % 20; gy = i / 20; }
    xc = ((float)gx + 0.5f) * st;
    yc = ((float)gy + 0.5f) * st;
}

// ---------------- K0: per-anchor s1m, geometry filter, init scratch ----------------
__global__ __launch_bounds__(256) void k_prep(
    const float* __restrict__ pred, const float* __restrict__ tgt,
    const int* __restrict__ insz,
    float* __restrict__ s1m, int* __restrict__ afilt,
    int* __restrict__ amg, int* __restrict__ selm, int* __restrict__ argm)
{
    __shared__ float gtb[MGT][4];
    __shared__ int   vld[MGT];
    int b = blockIdx.y, tid = threadIdx.x;
    float scale = (float)insz[0];
    if (tid < MGT) {
        const float* t = tgt + ((size_t)b * MGT + tid) * 5;
        vld[tid] = (t[0] >= 0.f) ? 1 : 0;
        gtb[tid][0] = t[1] * scale; gtb[tid][1] = t[2] * scale;
        gtb[tid][2] = t[3] * scale; gtb[tid][3] = t[4] * scale;
    }
    __syncthreads();
    int a = blockIdx.x * 256 + tid;
    if (a >= A_TOT) return;
    float xc, yc, st; anchor_of(a, xc, yc, st);
    size_t pb = ((size_t)b * A_TOT + a) * (5 + NCLS);
    float so = sigf(pred[pb + 4]);
    float s = 0.f;
    for (int c = 0; c < NCLS; ++c) {
        float p = sqrtf(sigf(pred[pb + 5 + c]) * so);
        s += fmaxf(log1pf(-p), -100.f);
    }
    size_t idx = (size_t)b * A_TOT + a;
    s1m[idx] = s;
    float cd = st * 1.5f;
    int f = 0;
    for (int m = 0; m < MGT; ++m) {
        if (!vld[m]) continue;
        float cl = xc - (gtb[m][0] - cd), cr = (gtb[m][0] + cd) - xc;
        float ct = yc - (gtb[m][1] - cd), cb = (gtb[m][1] + cd) - yc;
        if (fminf(fminf(cl, cr), fminf(ct, cb)) > 0.f) { f = 1; break; }
    }
    afilt[idx] = f; amg[idx] = 0; selm[idx] = 0; argm[idx] = 0;
}

// ---------------- K1: WMMA cost matrix + SimOTA dynamic-k assignment ----------------
__global__ __launch_bounds__(256) void k_assign(
    const float* __restrict__ pred, const float* __restrict__ tgt,
    const int* __restrict__ insz,
    const float* __restrict__ s1m, const int* __restrict__ afilt,
    int* __restrict__ amg, int* __restrict__ selm, int* __restrict__ argm)
{
    __shared__ float    gtbS[64][4];
    __shared__ int      cgS[64];
    __shared__ int      vS[64];
    __shared__ _Float16 Rl[64][100];     // R = log1mp - logp, classes padded to 96
    __shared__ float    costS[64][65];
    __shared__ float    iouS[64][65];
    __shared__ __align__(16) float pboxS[64][4];
    __shared__ float    s1mS[64];
    __shared__ int      afS[64];
    __shared__ float    sgoS[64];
    __shared__ float    xcS[64], ycS[64], cdS[64];

    int b = blockIdx.x, tid = threadIdx.x;
    float scale = (float)insz[0];
    if (tid < 64) {
        if (tid < MGT) {
            const float* t = tgt + ((size_t)b * MGT + tid) * 5;
            float c0 = t[0];
            vS[tid] = (c0 >= 0.f) ? 1 : 0;
            cgS[tid] = (int)fminf(fmaxf(c0, 0.f), (float)(NCLS - 1));
            gtbS[tid][0] = t[1] * scale; gtbS[tid][1] = t[2] * scale;
            gtbS[tid][2] = t[3] * scale; gtbS[tid][3] = t[4] * scale;
        } else {
            vS[tid] = 0; cgS[tid] = -1;
            gtbS[tid][0] = 0.f; gtbS[tid][1] = 0.f; gtbS[tid][2] = 1.f; gtbS[tid][3] = 1.f;
        }
    }
    // per-GT-row streaming top-k state (registers/scratch)
    float tc[TOPKN]; int ti[TOPKN]; float tu[TOPKN];
#pragma unroll
    for (int k = 0; k < TOPKN; ++k) { tc[k] = __builtin_inff(); ti[k] = 0; tu[k] = 0.f; }
    __syncthreads();

    int wave = tid >> 5, lane = tid & 31;
    int mt    = wave & 3;
    int lrow  = lane & 15;
    int koff0 = (lane & 16) ? 8 : 0;

    // One-hot A fragments are invariant across ALL chunks and N-tiles: build once.
    int cgm = cgS[mt * 16 + lrow];
    v16h afr[3];
#pragma unroll
    for (int ks = 0; ks < 3; ++ks) {
#pragma unroll
        for (int h = 0; h < 16; ++h) {
            int K = ks * 32 + koff0 + h + ((h >= 8) ? 8 : 0);
            afr[ks][h] = (_Float16)((cgm == K) ? 1.0f : 0.0f);
        }
    }

    for (int cb = 0; cb < NCHUNK; ++cb) {
        int base = cb * CHUNK;
        // ---- stage per-anchor chunk data ----
        if (tid < 64) {
            int a = base + tid;
            if (a < A_TOT) {
                size_t pb = ((size_t)b * A_TOT + a) * (5 + NCLS);
#if USE_ASYNC_LDS
                // async global->LDS copy of the 4 box floats (ASYNCcnt-tracked,
                // DWORD-aligned since the 85-float row is only 4B aligned)
#pragma unroll
                for (int q = 0; q < 4; ++q)
                    __builtin_amdgcn_global_load_async_to_lds_b32(
                        (__attribute__((address_space(1))) int*)(pred + pb + q),
                        (__attribute__((address_space(3))) int*)&pboxS[tid][q], 0, 0);
#else
                pboxS[tid][0] = pred[pb];     pboxS[tid][1] = pred[pb + 1];
                pboxS[tid][2] = pred[pb + 2]; pboxS[tid][3] = pred[pb + 3];
#endif
                sgoS[tid] = sigf(pred[pb + 4]);
                s1mS[tid] = s1m[(size_t)b * A_TOT + a];
                afS[tid]  = afilt[(size_t)b * A_TOT + a];
                float xc, yc, st; anchor_of(a, xc, yc, st);
                xcS[tid] = xc; ycS[tid] = yc; cdS[tid] = st * 1.5f;
                if (a + CHUNK < A_TOT)   // warm L2 for the next chunk (global_prefetch_b8)
                    __builtin_prefetch(&pred[((size_t)b * A_TOT + a + CHUNK) * (5 + NCLS)], 0, 1);
            } else {
                pboxS[tid][0] = 0.f; pboxS[tid][1] = 0.f; pboxS[tid][2] = 1.f; pboxS[tid][3] = 1.f;
                sgoS[tid] = 0.5f; s1mS[tid] = 0.f; afS[tid] = 0;
                xcS[tid] = 0.f; ycS[tid] = 0.f; cdS[tid] = 0.f;
            }
        }
#if USE_ASYNC_LDS
        __builtin_amdgcn_s_wait_asynccnt(0);
#endif
        __syncthreads();
        // ---- build R[a][c] tile (f16) ----
        for (int e = tid; e < CHUNK * 96; e += 256) {
            int al = e & 63, c = e >> 6;
            float r = 0.f;
            int a = base + al;
            if (c < NCLS && a < A_TOT) {
                float p  = sqrtf(sigf(pred[((size_t)b * A_TOT + a) * (5 + NCLS) + 5 + c]) * sgoS[al]);
                float lp = fmaxf(logf(p), -100.f);
                float l1 = fmaxf(log1pf(-p), -100.f);
                r = l1 - lp;
            }
            Rl[al][c] = (_Float16)r;
        }
        __syncthreads();
        // ---- WMMA: cls-cost one-hot GEMM, 16 tiles of 16x16 over K=96 ----
#pragma unroll
        for (int rep = 0; rep < 2; ++rep) {
            int at   = (wave >> 2) * 2 + rep;
            int colA = at * 16 + lrow;
            v8f acc = {};
#pragma unroll
            for (int ks = 0; ks < 3; ++ks) {
                v16h bf;
#pragma unroll
                for (int h = 0; h < 16; ++h) {
                    int K = ks * 32 + koff0 + h + ((h >= 8) ? 8 : 0);
                    bf[h] = Rl[colA][K];
                }
                acc = __builtin_amdgcn_wmma_f32_16x16x32_f16(
                    false, afr[ks], false, bf, (short)0, acc, false, false);
            }
            int Md  = mt * 16 + ((lane & 16) ? 8 : 0);
            int col = at * 16 + lrow;
#pragma unroll
            for (int v = 0; v < 8; ++v) costS[Md + v][col] = acc[v];
        }
        __syncthreads();
        // ---- elementwise finalize: IoU + penalties + masking ----
        for (int e = tid; e < CHUNK * 64; e += 256) {
            int mrow = e >> 6, col = e & 63;
            if (mrow >= MGT) continue;
            int a = base + col;
            if (a >= A_TOT) { costS[mrow][col] = __builtin_inff(); iouS[mrow][col] = 0.f; continue; }
            int rv = vS[mrow] & afS[col];
            float gx = gtbS[mrow][0], gy = gtbS[mrow][1], gw = gtbS[mrow][2], gh = gtbS[mrow][3];
            float cd = cdS[col], xc = xcS[col], yc = ycS[col];
            float cl = xc - (gx - cd), cr = (gx + cd) - xc;
            float ct = yc - (gy - cd), cbv = (gy + cd) - yc;
            int inc = vS[mrow] && (fminf(fminf(cl, cr), fminf(ct, cbv)) > 0.f);
            float px = pboxS[col][0], py = pboxS[col][1], pw = pboxS[col][2], ph = pboxS[col][3];
            float tlx = fmaxf(gx - gw * 0.5f, px - pw * 0.5f);
            float tly = fmaxf(gy - gh * 0.5f, py - ph * 0.5f);
            float brx = fminf(gx + gw * 0.5f, px + pw * 0.5f);
            float bry = fminf(gy + gh * 0.5f, py + ph * 0.5f);
            float iou = 0.f;
            if (rv && tlx < brx && tly < bry) {
                float ai = (brx - tlx) * (bry - tly);
                iou = ai / (gw * gh + pw * ph - ai);
            }
            float icost = -logf(iou + 1e-8f);
            float cval = costS[mrow][col] - s1mS[col] + 3.f * icost + (inc ? 0.f : 1e6f);
            costS[mrow][col] = rv ? cval : BIGC;
            iouS[mrow][col]  = iou;
        }
        __syncthreads();
        // ---- scans: rows keep top-10 cost/IoU, columns compute argmin over m ----
        if (tid < MGT && vS[tid]) {
            for (int col = 0; col < CHUNK; ++col) {
                float cv = costS[tid][col];
                if (cv < tc[TOPKN - 1]) {
                    int p = TOPKN - 1;
                    while (p > 0 && cv < tc[p - 1]) { tc[p] = tc[p - 1]; ti[p] = ti[p - 1]; --p; }
                    tc[p] = cv; ti[p] = base + col;
                }
                float uv = iouS[tid][col];
                if (uv > tu[TOPKN - 1]) {
                    int p = TOPKN - 1;
                    while (p > 0 && uv > tu[p - 1]) { tu[p] = tu[p - 1]; --p; }
                    tu[p] = uv;
                }
            }
        } else if (tid >= 64 && tid < 128) {
            int col = tid - 64, a = base + col;
            if (a < A_TOT) {
                float best = __builtin_inff(); int bm = 0;
                for (int m = 0; m < MGT; ++m) {
                    float cv = costS[m][col];
                    if (cv < best) { best = cv; bm = m; }
                }
                argm[(size_t)b * A_TOT + a] = bm;
            }
        }
        __syncthreads();
    }
    // ---- dynamic-k selection ----
    if (tid < MGT && vS[tid]) {
        float su = 0.f;
#pragma unroll
        for (int k = 0; k < TOPKN; ++k) su += tu[k];
        int dk = (int)su; if (dk < 1) dk = 1; if (dk > TOPKN) dk = TOPKN;
        for (int k = 0; k < dk; ++k) {
            int a = ti[k];
            atomicAdd(&amg[(size_t)b * A_TOT + a], 1);
            atomicExch(&selm[(size_t)b * A_TOT + a], tid);
        }
    }
}

// ---------------- K2: losses + deterministic block reduction ----------------
__global__ __launch_bounds__(256) void k_loss(
    const float* __restrict__ pred, const float* __restrict__ tgt,
    const int* __restrict__ insz,
    const int* __restrict__ amg, const int* __restrict__ selm, const int* __restrict__ argm,
    float* __restrict__ partial)
{
    __shared__ int   cg[MGT];
    __shared__ float gtb[MGT][4];
    __shared__ float red[256];
    int b = blockIdx.y, tid = threadIdx.x;
    float scale = (float)insz[0];
    if (tid < MGT) {
        const float* t = tgt + ((size_t)b * MGT + tid) * 5;
        cg[tid] = (int)fminf(fmaxf(t[0], 0.f), (float)(NCLS - 1));
        gtb[tid][0] = t[1] * scale; gtb[tid][1] = t[2] * scale;
        gtb[tid][2] = t[3] * scale; gtb[tid][3] = t[4] * scale;
    }
    __syncthreads();
    int a = blockIdx.x * 256 + tid;
    float obj = 0.f, cls = 0.f, box = 0.f, np = 0.f;
    if (a < A_TOT) {
        size_t pb = ((size_t)b * A_TOT + a) * (5 + NCLS);
        int cnt = amg[(size_t)b * A_TOT + a];
        float tf = (cnt > 0) ? 1.f : 0.f;
        obj = bcef(pred[pb + 4], tf);
        if (cnt > 0) {
            int mg = (cnt == 1) ? selm[(size_t)b * A_TOT + a] : argm[(size_t)b * A_TOT + a];
            int cc = cg[mg];
            for (int c = 0; c < NCLS; ++c) cls += bcef(pred[pb + 5 + c], (c == cc) ? 1.f : 0.f);
            const float eps = 1e-7f;
            float px = pred[pb], py = pred[pb + 1], pw = pred[pb + 2], ph = pred[pb + 3];
            float gx = gtb[mg][0], gy = gtb[mg][1], gw = gtb[mg][2], gh = gtb[mg][3];
            float px1 = px - pw * 0.5f, py1 = py - ph * 0.5f, px2 = px + pw * 0.5f, py2 = py + ph * 0.5f;
            float gx1 = gx - gw * 0.5f, gy1 = gy - gh * 0.5f, gx2 = gx + gw * 0.5f, gy2 = gy + gh * 0.5f;
            float iw = fmaxf(fminf(px2, gx2) - fmaxf(px1, gx1), 0.f);
            float ih = fmaxf(fminf(py2, gy2) - fmaxf(py1, gy1), 0.f);
            float inter = iw * ih;
            float uni = pw * ph + gw * gh - inter + eps;
            float iou = inter / uni;
            float cw = fmaxf(px2, gx2) - fminf(px1, gx1);
            float ch = fmaxf(py2, gy2) - fminf(py1, gy1);
            float c2 = cw * cw + ch * ch + eps;
            float rho2 = (gx - px) * (gx - px) + (gy - py) * (gy - py);
            float dv = atanf(gw / (gh + eps)) - atanf(pw / (ph + eps));
            float v = (4.f / (PI_F * PI_F)) * dv * dv;
            float alpha = v / (1.f - iou + v + eps);
            box = 1.f - iou + rho2 / c2 + alpha * v;
            np = 1.f;
        }
    }
    float vals[4] = { obj, cls, box, np };
    for (int q = 0; q < 4; ++q) {
        red[tid] = vals[q]; __syncthreads();
        for (int s = 128; s > 0; s >>= 1) {
            if (tid < s) red[tid] += red[tid + s];
            __syncthreads();
        }
        if (tid == 0) partial[((size_t)b * NXB + blockIdx.x) * 4 + q] = red[0];
        __syncthreads();
    }
}

// ---------------- K3: fixed-order final reduction ----------------
__global__ __launch_bounds__(256) void k_final(const float* __restrict__ partial,
                                               float* __restrict__ out, int B)
{
    __shared__ float img[64][4];
    int tid = threadIdx.x;
    if (tid < B && tid < 64) {
        float o = 0.f, c = 0.f, bx = 0.f, np = 0.f;
        for (int xb = 0; xb < NXB; ++xb) {
            const float* p = partial + ((size_t)tid * NXB + xb) * 4;
            o += p[0]; c += p[1]; bx += p[2]; np += p[3];
        }
        img[tid][0] = o / (float)A_TOT;
        img[tid][1] = (np > 0.f) ? c / (fmaxf(np, 1.f) * (float)NCLS) : 0.f;
        img[tid][2] = (np > 0.f) ? bx / fmaxf(np, 1.f) : 0.f;
        img[tid][3] = np;
    }
    __syncthreads();
    if (tid == 0) {
        float os = 0.f, cs = 0.f, bs = 0.f, fg = 0.f;
        for (int b = 0; b < B && b < 64; ++b) {
            os += img[b][0]; cs += img[b][1]; bs += img[b][2]; fg += img[b][3];
        }
        float total = (5.f * bs + os + cs) / fmaxf(fg, 1.f);
        out[0] = total; out[1] = bs; out[2] = os; out[3] = cs; out[4] = fg;
    }
}

extern "C" void kernel_launch(void* const* d_in, const int* in_sizes, int n_in,
                              void* d_out, int out_size, void* d_ws, size_t ws_size,
                              hipStream_t stream)
{
    const float* pred = (const float*)d_in[0];
    const float* tgt  = (const float*)d_in[1];
    const int*   insz = (const int*)d_in[2];
    int B = in_sizes[0] / (A_TOT * (5 + NCLS));

    char* ws = (char*)d_ws;
    size_t n = (size_t)B * A_TOT;
    float* s1m   = (float*)ws; ws += n * 4;
    int*   afilt = (int*)ws;   ws += n * 4;
    int*   amg   = (int*)ws;   ws += n * 4;
    int*   selm  = (int*)ws;   ws += n * 4;
    int*   argm  = (int*)ws;   ws += n * 4;
    float* partial = (float*)ws;

    dim3 g0(NXB, B);
    k_prep  <<<g0, 256, 0, stream>>>(pred, tgt, insz, s1m, afilt, amg, selm, argm);
    k_assign<<<B,  256, 0, stream>>>(pred, tgt, insz, s1m, afilt, amg, selm, argm);
    k_loss  <<<g0, 256, 0, stream>>>(pred, tgt, insz, amg, selm, argm, partial);
    k_final <<<1,  256, 0, stream>>>(partial, (float*)d_out, B);
}
